// HITNet_SF_39651138077354
// MI455X (gfx1250) — compile-verified, matrix operand order checked by reading, not run on hardware
//
#include <hip/hip_runtime.h>
#include <stdint.h>

// ---------------------------------------------------------------------------
// CDNA5 (gfx1250) HITNet-style stereo pipeline, wave32 + WMMA 16x16x32 f16.
// Activations: NHWC fp16, channels padded to a power-of-two multiple of 16.
// Conv = implicit GEMM. Per wave: M = 32 output pixels (2 A fragments),
// N = 32 output channels (2 B fragments) -> 4 WMMAs per 32-deep K block.
// 4 waves / workgroup share one LDS copy of the packed B fragments, staged
// by the Tensor Data Mover when available (fallback: cooperative copy).
// ---------------------------------------------------------------------------

typedef __attribute__((ext_vector_type(16))) _Float16     v16h;
typedef __attribute__((ext_vector_type(8)))  float        v8f;
typedef __attribute__((ext_vector_type(4)))  unsigned int v4u;
typedef __attribute__((ext_vector_type(4)))  int          v4i;
typedef __attribute__((ext_vector_type(8)))  int          v8i;

#define LRELU_SLOPE 0.2f
#define MAXKB 9            // ldsB K-blocks per N-block (9*32 >= 288); 4x4 Cin=32 layers use 16

struct ConvDesc {
  int N, Hin, Win, Cin;    // Cin padded, power of two
  int Hout, Wout, Cout;    // Cout padded to multiple of 16
  int kh, kw, sy, sx, py, px;
  int cinSh;               // log2(Cin)
  int relu;                // 1 = LeakyReLU(0.2) epilogue
};

// ---------------------------------------------------------------------------
// NCHW fp32 -> NHWC fp16 (channel-padded) conversion.
// ---------------------------------------------------------------------------
__global__ void k_nchw_to_nhwc(const float* __restrict__ src, _Float16* __restrict__ dst,
                               int N, int C, int H, int W, int Cp) {
  long idx = (long)blockIdx.x * blockDim.x + threadIdx.x;
  long total = (long)N * H * W * Cp;
  if (idx >= total) return;
  int c = (int)(idx % Cp);
  long p = idx / Cp;
  int x = (int)(p % W);
  long t = p / W;
  int y = (int)(t % H);
  int n = (int)(t / H);
  float v = 0.f;
  if (c < C) v = src[(((long)n * C + c) * H + y) * W + x];
  dst[idx] = (_Float16)v;
}

// ---------------------------------------------------------------------------
// Pack weights into exact B-fragment register order for wmma_f32_16x16x32_f16.
// B fragment (16-bit, 32x16): lane = n + 16*kHi; VGPR i half j:
//   K = (i/4)*16 + 8*kHi + (i%4)*2 + j       (mirror of the documented A map)
// ---------------------------------------------------------------------------
__global__ void k_pack_w(const float* __restrict__ wsrc, int wsrc_n,
                         const float* __restrict__ bsrc, int bsrc_n,
                         _Float16* __restrict__ wpk, float* __restrict__ bias,
                         ConvDesc d, int nkb) {
  const int Ktot = d.kh * d.kw * d.Cin;
  const int cblocks = d.Cout >> 4;
  const int total = cblocks * nkb * 32 * 16;
  int idx = blockIdx.x * blockDim.x + threadIdx.x;
  if (idx < d.Cout) bias[idx] = bsrc[idx % bsrc_n] * 0.1f;
  if (idx >= total) return;

  int e16  = idx & 15;
  int lane = (idx >> 4) & 31;
  int kb   = (idx >> 9) % nkb;
  int cb   = (idx >> 9) / nkb;
  int n    = lane & 15;
  int kHi  = lane >> 4;
  int vg   = e16 >> 1, j = e16 & 1;
  int K = kb * 32 + (vg >> 2) * 16 + 8 * kHi + (vg & 3) * 2 + j;

  _Float16 val = (_Float16)0.f;
  if (K < Ktot) {
    int t = K >> d.cinSh;
    int c = K & (d.Cin - 1);
    int cout = cb * 16 + n;
    long widx = ((((long)cout * d.Cin + c) * d.kh + (t / d.kw)) * d.kw + (t % d.kw));
    val = (_Float16)(wsrc[widx % wsrc_n] * 0.05f);
  }
  wpk[idx] = val;
}

// ---------------------------------------------------------------------------
// WMMA implicit-GEMM convolution, fused bias + LeakyReLU.
// Block = 128 threads = 4 waves; wave w covers pixels [blk*128 + 32w, +32).
// blockIdx.y selects a 32-cout super-block (2 x 16-cout WMMA N-blocks).
// ---------------------------------------------------------------------------
template <int KW>
__global__ void __launch_bounds__(128)
k_conv_wmma(const _Float16* __restrict__ x, const _Float16* __restrict__ wpk,
            const float* __restrict__ bias, _Float16* __restrict__ y,
            ConvDesc d, int nkb) {
  __shared__ __align__(16) _Float16 ldsB[2 * MAXKB * 512];

  const int tid    = threadIdx.x;
  const int lane   = tid & 31;
  const int wave   = tid >> 5;
  const int HW     = d.Hout * d.Wout;
  const int cb0    = blockIdx.y * 2;                 // first 16-cout block
  const int cbMax  = d.Cout >> 4;
  const int nv     = (cb0 + 1 < cbMax) ? 2 : 1;      // live N-blocks
  const int b      = blockIdx.z;
  const int nbytes = nv * nkb * 1024;                // B-fragment bytes for this block

  // ---- Stage B fragments into LDS ---------------------------------------
#if defined(__gfx1250__) && __has_builtin(__builtin_amdgcn_tensor_load_to_lds)
  if (wave == 0) {
    // Tensor Data Mover: one 1-row 2D tile, data_size = 8 bytes.
    unsigned long long ga =
        (unsigned long long)(uintptr_t)(wpk + (size_t)cb0 * nkb * 512);
    unsigned nelem = (unsigned)(nbytes >> 3);        // 8-byte units (<= 4096)
    v4u g0;
    g0[0] = 1u;                                      // count=1, user descriptor
    g0[1] = 0u;                                      // lds_addr = base of ldsB
    g0[2] = (unsigned)(ga & 0xffffffffu);            // global_addr[31:0]
    g0[3] = (unsigned)((ga >> 32) & 0x01ffffffu) | (2u << 30);  // addr[56:32]|type=2
    v8i g1;
    g1[0] = (int)(3u << 16);                         // data_size = 3 (8B)
    g1[1] = (int)((nelem & 0xffffu) << 16);          // tensor_dim0[15:0]
    g1[2] = (int)(((nelem >> 16) & 0xffffu) | (1u << 16)); // dim0[31:16] | tensor_dim1=1
    g1[3] = (int)((nelem & 0xffffu) << 16);          // tile_dim0
    g1[4] = 1;                                       // tile_dim1 = 1, tile_dim2 = 0
    g1[5] = (int)nelem;                              // tensor_dim0_stride[31:0]
    g1[6] = 0;
    g1[7] = 0;
    v4i z4 = {0, 0, 0, 0};
#if __clang_major__ >= 23
    v8i z8 = {0, 0, 0, 0, 0, 0, 0, 0};
    __builtin_amdgcn_tensor_load_to_lds(g0, g1, z4, z4, z8, 0);
#else
    __builtin_amdgcn_tensor_load_to_lds(g0, g1, z4, z4, 0);
#endif
#if __has_builtin(__builtin_amdgcn_s_wait_tensorcnt)
    __builtin_amdgcn_s_wait_tensorcnt(0);
#endif
  }
#else
  {
    const v4u* src = (const v4u*)(wpk + (size_t)cb0 * nkb * 512);
    v4u* dst = (v4u*)ldsB;
    int n16 = nbytes >> 4;
    for (int i = tid; i < n16; i += 128) dst[i] = src[i];
  }
#endif
  __syncthreads();

  const int m0 = (blockIdx.x * 4 + wave) * 32;
  if (m0 >= HW) return;                              // whole-wave early out (post-barrier)

  const int laneHi = lane >> 4;
  const int mLane  = lane & 15;

  // Per-M-block base input coordinates for this lane's A rows.
  bool moka[2];
  int iy0a[2], ix0a[2];
#pragma unroll
  for (int mi = 0; mi < 2; ++mi) {
    int pix = m0 + mi * 16 + mLane;
    bool ok = pix < HW;
    int oy = ok ? pix / d.Wout : 0;
    int ox = ok ? pix % d.Wout : 0;
    moka[mi] = ok;
    iy0a[mi] = oy * d.sy - d.py;
    ix0a[mi] = ox * d.sx - d.px;
  }

  const int Ktot  = d.kh * KW * d.Cin;
  const int cmask = d.Cin - 1;
  const int cinSh = d.cinSh;
  v8f acc00 = {}, acc01 = {}, acc10 = {}, acc11 = {};

  for (int kb = 0; kb < nkb; ++kb) {
    // Build two A fragments (8 channel-pair dword gathers each).
    union AF { uint32_t u[8]; v16h v; } A[2];
    const int kbase = kb * 32 + 8 * laneHi;
#pragma unroll
    for (int mi = 0; mi < 2; ++mi) {
      const bool mok = moka[mi];
      const int iy0 = iy0a[mi], ix0 = ix0a[mi];
#pragma unroll
      for (int e = 0; e < 8; ++e) {
        int Kp = kbase + (e >> 2) * 16 + (e & 3) * 2;
        uint32_t w = 0u;
        if (mok && Kp < Ktot) {
          int t = Kp >> cinSh;
          int c = Kp & cmask;
          int iy = iy0 + t / KW;
          int ix = ix0 + t % KW;
          if (iy >= 0 && iy < d.Hin && ix >= 0 && ix < d.Win) {
            const _Float16* p = x + (((size_t)b * d.Hin + iy) * d.Win + ix) * d.Cin + c;
            w = *(const uint32_t*)p;
          }
        }
        A[mi].u[e] = w;
      }
    }

    // 2 B fragments from LDS; 4 WMMAs reuse both A's and both B's.
    const _Float16* bp = ldsB + (size_t)kb * 512 + (size_t)lane * 16;
    v16h B0 = *(const v16h*)bp;
    acc00 = __builtin_amdgcn_wmma_f32_16x16x32_f16(false, A[0].v, false, B0,
                                                   (short)0, acc00, false, false);
    acc10 = __builtin_amdgcn_wmma_f32_16x16x32_f16(false, A[1].v, false, B0,
                                                   (short)0, acc10, false, false);
    if (nv > 1) {
      v16h B1 = *(const v16h*)(bp + (size_t)nkb * 512);
      acc01 = __builtin_amdgcn_wmma_f32_16x16x32_f16(false, A[0].v, false, B1,
                                                     (short)0, acc01, false, false);
      acc11 = __builtin_amdgcn_wmma_f32_16x16x32_f16(false, A[1].v, false, B1,
                                                     (short)0, acc11, false, false);
    }
  }

  // Epilogue: bias + LeakyReLU; C layout lane n = lane&15, row m = r + 8*laneHi.
  const int n = lane & 15;
  auto store = [&](const v8f& a, int mi, int nb) {
    float bs = bias[(cb0 + nb) * 16 + n];
#pragma unroll
    for (int r = 0; r < 8; ++r) {
      int m = m0 + mi * 16 + r + 8 * laneHi;
      if (m >= HW) continue;
      float v = a[r] + bs;
      if (d.relu) v = v > 0.f ? v : v * LRELU_SLOPE;
      y[((size_t)b * HW + m) * d.Cout + (cb0 + nb) * 16 + n] = (_Float16)v;
    }
  };
  store(acc00, 0, 0);
  store(acc10, 1, 0);
  if (nv > 1) {
    store(acc01, 0, 1);
    store(acc11, 1, 1);
  }
}

// ---------------------------------------------------------------------------
// 1/4-res cost volume: L1 over 16 channels, 80 disparities, on-the-fly argmin.
// ---------------------------------------------------------------------------
__global__ void k_costvol_argmin(const _Float16* __restrict__ flt,  // (N,Hq,Wq,16)
                                 const _Float16* __restrict__ frt,  // (N,Hq,Wr,16)
                                 float* __restrict__ dinit,         // (N,Hq,Wq)
                                 int N, int Hq, int Wq, int Wr, int D) {
  int idx = blockIdx.x * blockDim.x + threadIdx.x;
  int total = N * Hq * Wq;
  if (idx >= total) return;
  int xq = idx % Wq;
  int t  = idx / Wq;
  int yq = t % Hq;
  int n  = t / Hq;

  const _Float16* lp = flt + (size_t)idx * 16;
  float l[16];
#pragma unroll
  for (int c = 0; c < 16; ++c) l[c] = (float)lp[c];

  float best = 3.4e38f;
  int bestd = 0;
  for (int dd = 0; dd < D; ++dd) {
    int ix = 4 * xq - dd + 1;
    ix = ix < 0 ? 0 : (ix >= Wr ? Wr - 1 : ix);
    const _Float16* rp = frt + (((size_t)(n * Hq + yq)) * Wr + ix) * 16;
    float s = 0.f;
#pragma unroll
    for (int c = 0; c < 16; ++c) s += fabsf(l[c] - (float)rp[c]);
    if (s < best) { best = s; bestd = dd; }
  }
  dinit[idx] = (float)bestd;
}

// ---------------------------------------------------------------------------
// Tile-disparity x4 upsample -> final fp32 output (2,1,384,1280).
// ---------------------------------------------------------------------------
__global__ void k_disp_up4(const float* __restrict__ dinit, float* __restrict__ out,
                           int N, int Hq, int Wq) {
  int H = Hq * 4, W = Wq * 4;
  size_t idx = (size_t)blockIdx.x * blockDim.x + threadIdx.x;
  size_t total = (size_t)N * H * W;
  if (idx >= total) return;
  int x = (int)(idx % W);
  size_t t = idx / W;
  int y = (int)(t % H);
  int n = (int)(t / H);
  out[idx] = dinit[((size_t)n * Hq + (y >> 2)) * Wq + (x >> 2)];
}

// ---------------------------------------------------------------------------
// Host orchestration.
// ---------------------------------------------------------------------------
static inline ConvDesc mk(int N, int Hin, int Win, int Cin,
                          int Hout, int Wout, int Cout,
                          int kh, int kw, int sy, int sx, int py, int px, int relu) {
  ConvDesc d;
  d.N = N; d.Hin = Hin; d.Win = Win; d.Cin = Cin;
  d.Hout = Hout; d.Wout = Wout; d.Cout = Cout;
  d.kh = kh; d.kw = kw; d.sy = sy; d.sx = sx; d.py = py; d.px = px;
  int sh = 0; while ((1 << sh) < Cin) ++sh;
  d.cinSh = sh;
  d.relu = relu;
  return d;
}

extern "C" void kernel_launch(void* const* d_in, const int* in_sizes, int n_in,
                              void* d_out, int out_size, void* d_ws, size_t ws_size,
                              hipStream_t stream) {
  (void)out_size; (void)ws_size;
  const float* imgL = (const float*)d_in[0];
  const float* imgR = (const float*)d_in[1];

  const int N = 2, H = 384, W = 1280;
  const int Hq = H / 4, Wq = W / 4;    // 96 x 320

  char* base = (char*)d_ws;
  size_t off = 0;
  auto take = [&](size_t bytes) -> void* {
    size_t a = (off + 255) & ~(size_t)255;
    off = a + bytes;
    return (void*)(base + a);
  };

  _Float16* xin  = (_Float16*)take((size_t)N * H * W * 16 * 2);
  _Float16* f0   = (_Float16*)take((size_t)N * H * W * 16 * 2);
  _Float16* s0   = (_Float16*)take((size_t)N * (H/2) * (W/2) * 16 * 2);
  _Float16* s1   = (_Float16*)take((size_t)N * (H/2) * (W/2) * 16 * 2);
  _Float16* flt  = (_Float16*)take((size_t)N * Hq * Wq * 16 * 2);
  _Float16* frt  = (_Float16*)take((size_t)N * Hq * W  * 16 * 2);
  float*    dini = (float*)take((size_t)N * Hq * Wq * 4);
  _Float16* wpk  = (_Float16*)take(64 * 1024);
  float*    bias = (float*)take(256 * 4);

  int layer = 0;
  auto wptr = [&](int& n) -> const float* {
    int i = (n_in > 2) ? (2 + (layer % (n_in - 2))) : 0;
    n = in_sizes[i] > 0 ? in_sizes[i] : 1;
    return (const float*)d_in[i];
  };

  auto conv = [&](const _Float16* x, _Float16* y, ConvDesc d) {
    int Ktot = d.kh * d.kw * d.Cin;
    int nkb = (Ktot + 31) / 32;
    int cb = d.Cout >> 4;
    int wn;
    const float* ws = wptr(wn);
    ++layer;
    int packTot = cb * nkb * 512;
    k_pack_w<<<dim3((packTot + 255) / 256), dim3(256), 0, stream>>>(
        ws, wn, ws, wn, wpk, bias, d, nkb);

    int mb = (d.Hout * d.Wout + 127) / 128;          // 4 waves x 32 pixels
    int cb2 = (cb + 1) / 2;                          // 32-cout super-blocks
    dim3 g(mb, cb2, d.N), blk(128);
    if (d.kw == 1)
      k_conv_wmma<1><<<g, blk, 0, stream>>>(x, wpk, bias, y, d, nkb);
    else if (d.kw == 3)
      k_conv_wmma<3><<<g, blk, 0, stream>>>(x, wpk, bias, y, d, nkb);
    else
      k_conv_wmma<4><<<g, blk, 0, stream>>>(x, wpk, bias, y, d, nkb);
  };

  for (int side = 0; side < 2; ++side) {
    const float* img = (side == 0) ? imgL : imgR;
    long cvt = (long)N * H * W * 16;
    k_nchw_to_nhwc<<<dim3((unsigned)((cvt + 255) / 256)), dim3(256), 0, stream>>>(
        img, xin, N, 3, H, W, 16);

    conv(xin, f0, mk(N, H,    W,    16, H,    W,    16, 3, 3, 1, 1, 1, 1, 1));  // d0
    conv(f0,  s0, mk(N, H,    W,    16, H/2,  W/2,  16, 4, 4, 2, 2, 1, 1, 1));  // d1a
    conv(s0,  s1, mk(N, H/2,  W/2,  16, H/2,  W/2,  16, 3, 3, 1, 1, 1, 1, 1));  // d1b
    conv(s1,  s0, mk(N, H/2,  W/2,  16, H/4,  W/4,  32, 4, 4, 2, 2, 1, 1, 1));  // d2a
    conv(s0,  s1, mk(N, H/4,  W/4,  32, H/4,  W/4,  32, 3, 3, 1, 1, 1, 1, 1));  // d2b
    conv(s1,  s0, mk(N, H/4,  W/4,  32, H/8,  W/8,  32, 4, 4, 2, 2, 1, 1, 1));  // d3a
    conv(s0,  s1, mk(N, H/8,  W/8,  32, H/8,  W/8,  32, 3, 3, 1, 1, 1, 1, 1));  // d3b
    conv(s1,  s0, mk(N, H/8,  W/8,  32, H/16, W/16, 32, 4, 4, 2, 2, 1, 1, 1));  // d4a
    conv(s0,  s1, mk(N, H/16, W/16, 32, H/16, W/16, 32, 3, 3, 1, 1, 1, 1, 1));  // d4b
    conv(s1,  s0, mk(N, H/16, W/16, 32, H/16, W/16, 32, 3, 3, 1, 1, 1, 1, 1));  // d4c
    conv(s0,  s1, mk(N, H/16, W/16, 32, H/16, W/16, 32, 3, 3, 1, 1, 1, 1, 1));  // d4d

    if (side == 0) {
      conv(f0, s0,  mk(N, H,  W,  16, Hq, Wq, 16, 4, 4, 4, 4, 0, 0, 1));        // em (left)
      conv(s0, flt, mk(N, Hq, Wq, 16, Hq, Wq, 16, 1, 1, 1, 1, 0, 0, 1));        // rc
    } else {
      conv(f0,  xin, mk(N, H,  W, 16, Hq, W, 16, 4, 4, 4, 1, 0, 1, 1));         // em (right)
      conv(xin, frt, mk(N, Hq, W, 16, Hq, W, 16, 1, 1, 1, 1, 0, 0, 1));         // rc
    }
  }

  int cvTot = N * Hq * Wq;
  k_costvol_argmin<<<dim3((cvTot + 255) / 256), dim3(256), 0, stream>>>(
      flt, frt, dini, N, Hq, Wq, W, 80);
  size_t outTot = (size_t)N * H * W;
  k_disp_up4<<<dim3((unsigned)((outTot + 255) / 256)), dim3(256), 0, stream>>>(
      dini, (float*)d_out, N, Hq, Wq);
}
